// LatentAttention_60730837566202
// MI455X (gfx1250) — compile-verified, matrix operand order checked by reading, not run on hardware
//
#include <hip/hip_runtime.h>

typedef __attribute__((ext_vector_type(2)))  float    v2f;
typedef __attribute__((ext_vector_type(8)))  float    v8f;
typedef __attribute__((ext_vector_type(8)))  _Float16 v8h;
typedef __attribute__((ext_vector_type(16))) _Float16 v16h;

#define BQ 32
#define LQ 1024
#define HQ 8
#define HDQ 8
#define DQ 64

static __device__ __forceinline__ v8f wmma_f32_4(v2f a, v2f b, v8f c) {
  // D = A(16x4 f32) * B(4x16 f32) + C(16x16 f32)
  return __builtin_amdgcn_wmma_f32_16x16x4_f32(false, a, false, b, (short)0, c, false, false);
}

// -------------------------------------------------------------------------
// Kernel 1: QKV projection.  z[32768,64] @ w_qkv[64,192] + b -> scatter into
//   qws [B,H,L,8] f32, kws [B,H,L,8] f32, vws [B,H,8,L] f16 (transposed V).
// One wave computes one 16x16 tile via 16 chained v_wmma_f32_16x16x4_f32.
// -------------------------------------------------------------------------
__global__ void qkv_kernel(const float* __restrict__ z, const float* __restrict__ w,
                           const float* __restrict__ bias,
                           float* __restrict__ qws, float* __restrict__ kws,
                           _Float16* __restrict__ vws) {
  const int lane = threadIdx.x;                 // 0..31
  const int wid  = blockIdx.x * blockDim.y + threadIdx.y;
  const int MT   = (BQ * LQ) / 16;              // 2048 row tiles
  const int mt   = wid % MT;
  const int nt   = wid / MT;                    // 0..11 (192/16)
  if (nt >= 12) return;
  const int n  = lane & 15;                     // N col (B) / M row (A) within tile
  const int g  = lane >> 4;                     // lane half
  const int r0 = mt * 16;
  const int n0 = nt * 16;

  v8f acc = {};
#pragma unroll
  for (int s = 0; s < 16; ++s) {
    const int ks = 4 * s;
    // A frag: lanes 0-15 -> K=ks,ks+1 ; lanes 16-31 -> K=ks+2,ks+3
    v2f a = *(const v2f*)(z + (size_t)(r0 + n) * 64 + ks + 2 * g);
    v2f b;
    b.x = w[(size_t)(ks + 2 * g    ) * 192 + n0 + n];
    b.y = w[(size_t)(ks + 2 * g + 1) * 192 + n0 + n];
    acc = wmma_f32_4(a, b, acc);
  }

  const float bj = bias[n0 + n];
  const int j  = n0 + n;
  const int d  = j & 63;
  const int h  = d >> 3;
  const int hd = d & 7;
#pragma unroll
  for (int i = 0; i < 8; ++i) {
    const float cv = acc[i] + bj;
    const int r  = r0 + i + 8 * g;              // global row = b*L + l
    const int bb = r >> 10;
    const int l  = r & 1023;
    const int bh = bb * HQ + h;
    if (nt < 4)      qws[((size_t)bh * LQ + l) * HDQ + hd] = cv;
    else if (nt < 8) kws[((size_t)bh * LQ + l) * HDQ + hd] = cv;
    else             vws[((size_t)bh * HDQ + hd) * LQ + l] = (_Float16)cv;
  }
}

// -------------------------------------------------------------------------
// Kernel 2: causal flash attention, one wave per (b,h, 16-query tile).
//   Next K/V 32-row slab is async-copied into per-wave LDS double buffers
//   (global_load_async_to_lds_b128, ASYNCcnt) while the current slab runs:
//   S^T = K @ Q^T via v_wmma_f32_16x16x4_f32 (k rows in VGPRs, q across lanes)
//   online softmax: in-lane reduce over 8 VGPRs + xor-16 lane swap
//   O^T += V^T @ P^T via v_wmma_f32_16x16x32_f16 (P^T packs in-lane into B frag)
// -------------------------------------------------------------------------
#define KSLAB 1024                 // 32 rows x 8 f32
#define VSLAB 512                  // 8 rows  x 32 f16
#define WAVE_LDS (2 * KSLAB + 2 * VSLAB)   // 3 KB per wave, double buffered

static __device__ __forceinline__ unsigned lds_off(const void* p) {
  // generic pointer to LDS: low 32 bits are the LDS byte address on gfx1250
  return (unsigned)(size_t)p;
}

__global__ void attn_kernel(const float* __restrict__ qws, const float* __restrict__ kws,
                            const _Float16* __restrict__ vws, float* __restrict__ aws) {
  __shared__ __align__(16) char smem[8 * WAVE_LDS];
  const int lane = threadIdx.x;
  const int wid  = blockIdx.x * blockDim.y + threadIdx.y;
  const int qt = wid & 63;                      // query tile 0..63
  const int bh = wid >> 6;                      // 0..255
  if (bh >= BQ * HQ) return;
  const int n  = lane & 15;
  const int g  = lane >> 4;
  const int q0 = qt * 16;
  const float scale = 0.35355339059327373f;     // 1/sqrt(8), folded into Q

  char* mybuf = smem + threadIdx.y * WAVE_LDS;  // [K0|K1|V0|V1]

  // Q^T fragments (B operand): step s covers head dims 4s..4s+3
  v2f qf0 = *(const v2f*)(qws + ((size_t)bh * LQ + q0 + n) * HDQ + 2 * g);
  v2f qf1 = *(const v2f*)(qws + ((size_t)bh * LQ + q0 + n) * HDQ + 4 + 2 * g);
  qf0 *= scale; qf1 *= scale;

  const int nkb = qt / 2 + 1;                   // causal: k <= q0+15

  // issue async copy of 32-row K/V slab `kb` into buffer kb&1 (3 ops)
  auto issue_slab = [&](int kb) {
    const int kbase = kb * 32;
    char* kslab = mybuf + (kb & 1) * KSLAB;
    char* vslab = mybuf + 2 * KSLAB + (kb & 1) * VSLAB;
    unsigned kldst = lds_off(kslab) + 16u * lane;
    unsigned kgoff = (unsigned)((bh * LQ + kbase) * 32) + 16u * lane;  // bytes
    asm volatile("global_load_async_to_lds_b128 %0, %1, %2 offset:0"
                 :: "v"(kldst), "v"(kgoff), "s"(kws) : "memory");
    asm volatile("global_load_async_to_lds_b128 %0, %1, %2 offset:512"
                 :: "v"(kldst), "v"(kgoff), "s"(kws) : "memory");
    unsigned vldst = lds_off(vslab) + 16u * lane;
    unsigned vgoff = (unsigned)((((bh * HDQ + (lane >> 2)) * LQ + kbase) * 2) +
                                16 * (lane & 3));                      // bytes
    asm volatile("global_load_async_to_lds_b128 %0, %1, %2 offset:0"
                 :: "v"(vldst), "v"(vgoff), "s"(vws) : "memory");
  };

  issue_slab(0);

  float mrun = -1e30f, lrun = 0.0f;
  v8f ot = {};
  for (int kb = 0; kb < nkb; ++kb) {
    const int kbase = kb * 32;
    const char* kslab = mybuf + (kb & 1) * KSLAB;
    const char* vslab = mybuf + 2 * KSLAB + (kb & 1) * VSLAB;

    if (kb + 1 < nkb) {
      issue_slab(kb + 1);
      // slab kb's 3 ops complete before the 3 just issued (in-order loads)
      asm volatile("s_wait_asynccnt 0x3" ::: "memory");
    } else {
      asm volatile("s_wait_asynccnt 0x0" ::: "memory");
    }

    v8f st[2];
#pragma unroll
    for (int t = 0; t < 2; ++t) {
      const int kk = kbase + 16 * t;
      v2f kf0 = *(const v2f*)(kslab + (16 * t + n) * 32 + 8 * g);       // hd {2g,2g+1}
      v2f kf1 = *(const v2f*)(kslab + (16 * t + n) * 32 + 16 + 8 * g);  // hd {4+2g,...}
      v8f s = {};
      s = wmma_f32_4(kf0, qf0, s);
      s = wmma_f32_4(kf1, qf1, s);
      if (kk + 15 > q0) {                       // tile touches the diagonal
#pragma unroll
        for (int i = 0; i < 8; ++i) {
          const int kidx = kk + i + 8 * g;      // C layout: M = vgpr + 8*lanehalf
          if (kidx > q0 + n) s[i] = -1e30f;
        }
      }
      st[t] = s;
    }

    // ---- online softmax over the 32-k slab (per q column = per lane) ----
    float tmax = -1e30f;
#pragma unroll
    for (int i = 0; i < 8; ++i) { tmax = fmaxf(tmax, st[0][i]); tmax = fmaxf(tmax, st[1][i]); }
    tmax = fmaxf(tmax, __shfl_xor(tmax, 16, 32));
    const float mnew  = fmaxf(mrun, tmax);
    const float alpha = __expf(mrun - mnew);
    float ls = 0.0f;
    v16h pf;                                    // P^T as B-fragment, pure in-lane pack
#pragma unroll
    for (int i = 0; i < 8; ++i) {
      const float p0 = __expf(st[0][i] - mnew);
      const float p1 = __expf(st[1][i] - mnew);
      ls += p0 + p1;
      pf[i]     = (_Float16)p0;                 // halves 0-7  = k tile 0
      pf[i + 8] = (_Float16)p1;                 // halves 8-15 = k tile 1
    }
    ls += __shfl_xor(ls, 16, 32);
    lrun = lrun * alpha + ls;
    mrun = mnew;
#pragma unroll
    for (int i = 0; i < 8; ++i) ot[i] *= alpha; // per-lane rescale of O^T

    // ---- V^T fragment (A operand) from LDS: rows 8-15 are zero padding ----
    v16h vf = {};
    if (n < 8) {
      const char* vrow = vslab + n * 64;        // 32 f16 of head-dim row n
      v8h lo = *(const v8h*)(vrow + (g ? 16 : 0));   // K {0..7} / {8..15}
      v8h hi = *(const v8h*)(vrow + (g ? 48 : 32));  // K {16..23} / {24..31}
      vf = __builtin_shufflevector(lo, hi, 0,1,2,3,4,5,6,7,8,9,10,11,12,13,14,15);
    }
    ot = __builtin_amdgcn_wmma_f32_16x16x32_f16(false, vf, false, pf, (short)0, ot,
                                                false, false);
  }

  // ---- epilogue: O^T rows 0-7 live in lanes 0-15, VGPRs 0-7 ----
  const float inv = 1.0f / lrun;
  if (lane < 16) {
    float* orow = aws + ((size_t)((bh >> 3) * LQ + q0 + lane)) * DQ + (bh & 7) * HDQ;
#pragma unroll
    for (int i = 0; i < 4; ++i) {
      v2f o; o.x = ot[2 * i] * inv; o.y = ot[2 * i + 1] * inv;
      *(v2f*)(orow + 2 * i) = o;
    }
  }
}

// -------------------------------------------------------------------------
// Kernel 3: output projection. aws[32768,64] @ w_proj[64,64] + b -> d_out.
// -------------------------------------------------------------------------
__global__ void proj_kernel(const float* __restrict__ a, const float* __restrict__ w,
                            const float* __restrict__ bias, float* __restrict__ out) {
  const int lane = threadIdx.x;
  const int wid  = blockIdx.x * blockDim.y + threadIdx.y;
  const int MT   = (BQ * LQ) / 16;
  const int mt   = wid % MT;
  const int nt   = wid / MT;                    // 0..3
  if (nt >= 4) return;
  const int n  = lane & 15;
  const int g  = lane >> 4;
  const int r0 = mt * 16, n0 = nt * 16;

  v8f acc = {};
#pragma unroll
  for (int s = 0; s < 16; ++s) {
    const int ks = 4 * s;
    v2f av = *(const v2f*)(a + (size_t)(r0 + n) * 64 + ks + 2 * g);
    v2f bv;
    bv.x = w[(size_t)(ks + 2 * g    ) * 64 + n0 + n];
    bv.y = w[(size_t)(ks + 2 * g + 1) * 64 + n0 + n];
    acc = wmma_f32_4(av, bv, acc);
  }
  const float bj = bias[n0 + n];
#pragma unroll
  for (int i = 0; i < 8; ++i)
    out[(size_t)(r0 + i + 8 * g) * 64 + n0 + n] = acc[i] + bj;
}

extern "C" void kernel_launch(void* const* d_in, const int* in_sizes, int n_in,
                              void* d_out, int out_size, void* d_ws, size_t ws_size,
                              hipStream_t stream) {
  (void)in_sizes; (void)n_in; (void)out_size; (void)ws_size;
  const float* z      = (const float*)d_in[0];
  const float* w_qkv  = (const float*)d_in[1];
  const float* b_qkv  = (const float*)d_in[2];
  const float* w_proj = (const float*)d_in[3];
  const float* b_proj = (const float*)d_in[4];
  float* out = (float*)d_out;

  const size_t NE = (size_t)BQ * HQ * LQ * HDQ;         // 2 Mi elements
  float*    qws = (float*)d_ws;                          // 8 MB
  float*    kws = qws + NE;                              // 8 MB
  _Float16* vws = (_Float16*)(kws + NE);                 // 4 MB (V^T f16)
  float*    aws = (float*)(vws + NE);                    // 8 MB ([B,L,H*HD])

  dim3 blk(32, 8);                                       // 8 wave32 per block
  qkv_kernel <<<3072, blk, 0, stream>>>(z, w_qkv, b_qkv, qws, kws, vws); // 24576 waves
  attn_kernel<<<2048, blk, 0, stream>>>(qws, kws, vws, aws);             // 16384 waves
  proj_kernel<<<1024, blk, 0, stream>>>(aws, w_proj, b_proj, out);       // 8192 waves
}
